// TransformerBlock_81046032876007
// MI455X (gfx1250) — matrix operand order repaired and checked by reading, not compiled, not run on hardware
//
#include <hip/hip_runtime.h>
#include <hip/hip_bf16.h>
#include <math.h>

// Problem constants (match reference)
#define BB   2
#define TT   2048
#define DD   768
#define HH   12
#define DKK  64
#define EE   8
#define DFF_ 3072
#define BT_  (BB * TT)
#define NEGF (-1e30f)

typedef __bf16 bf16;
typedef bf16  bf16x16 __attribute__((ext_vector_type(16)));
typedef bf16  bf16x8  __attribute__((ext_vector_type(8)));
typedef float v8f     __attribute__((ext_vector_type(8)));

// ---------------------------------------------------------------------------
// Optional CDNA5 async global->LDS copy (ASYNCcnt-tracked). Falls back to
// plain vector copies when the builtin is unavailable on this toolchain.
// The builtin on this toolchain takes pointers to a 16B int vector; the
// global source is generic, the LDS destination is cast to address_space(3)
// (AS3 -> generic also converts implicitly if the param is generic).
// ---------------------------------------------------------------------------
#if __has_builtin(__builtin_amdgcn_global_load_async_to_lds_b128) && \
    __has_builtin(__builtin_amdgcn_s_wait_asynccnt)
#define HAVE_ASYNC_LDS 1
typedef int v4i_vs __attribute__((__vector_size__(4 * sizeof(int))));
typedef __attribute__((address_space(3))) v4i_vs v4i_lds;
__device__ __forceinline__ void async_copy_b128(const bf16* g, bf16* l) {
  __builtin_amdgcn_global_load_async_to_lds_b128(
      (v4i_vs*)g, (v4i_lds*)l, 0, 0);
}
__device__ __forceinline__ void async_wait0() {
  __builtin_amdgcn_s_wait_asynccnt(0);
}
#else
#define HAVE_ASYNC_LDS 0
__device__ __forceinline__ void async_copy_b128(const bf16* g, bf16* l) {
  *(bf16x8*)l = *(const bf16x8*)g;
}
__device__ __forceinline__ void async_wait0() {}
#endif

// ---------------------------------------------------------------------------
// WMMA helper: D = A(16x32 bf16) * B(32x16 bf16) + C(16x16 f32)
// ---------------------------------------------------------------------------
__device__ __forceinline__ v8f wmma_bf16(bf16x16 a, bf16x16 b, v8f c) {
  return __builtin_amdgcn_wmma_f32_16x16x32_bf16(
      /*neg_a=*/false, a, /*neg_b=*/false, b,
      /*c_mod=*/(short)0, c, /*reuse_a=*/false, /*reuse_b=*/false);
}

// A-fragment (16x32 bf16) from an LDS tile `s` (row-major, leading dim 32).
// Contiguous k order (same as B frags): lanes 0-15: M=lane, K=0..15; lanes 16-31: M=lane-16,
// K=16..31.  16 contiguous elements -> two b128 loads.
__device__ __forceinline__ bf16x16 load_a_frag(const bf16* s, int lane) {
  int m  = lane & 15;
  int ko = (lane < 16) ? 0 : 16;
  const bf16x8* p = (const bf16x8*)(s + m * 32 + ko);
  bf16x8 lo = p[0];
  bf16x8 hi = p[1];  // +8 elements
  bf16x16 a;
#pragma unroll
  for (int j = 0; j < 8; ++j) { a[j] = lo[j]; a[j + 8] = hi[j]; }
  return a;
}

// B-fragment (32x16 bf16) from a TRANSPOSED LDS tile sT[n][k] (ld 32), columns
// [ncol, ncol+16).  Lane holds col n = ncol + (lane&15); element j -> K =
// (lane<16?0:16)+j, contiguous in sT -> two b128 loads.
__device__ __forceinline__ bf16x16 load_bT_frag(const bf16* sT, int ncol, int lane) {
  int n   = ncol + (lane & 15);
  int ko2 = (lane < 16) ? 0 : 16;
  const bf16x8* p = (const bf16x8*)(sT + n * 32 + ko2);
  bf16x8 lo = p[0];
  bf16x8 hi = p[1];
  bf16x16 b;
#pragma unroll
  for (int j = 0; j < 8; ++j) { b[j] = lo[j]; b[j + 8] = hi[j]; }
  return b;
}

// ---------------------------------------------------------------------------
// WMMA GEMM: C[M,N] = epilogue(A[M,K](bf16) @ B[K,N](f32->bf16) (+bias))
// Block tile 256x64x32, 256 thr = 8 waves; wave w owns rows [w*32, w*32+32)
// (two A-frags, 8 accumulators) -> 64 WMMAs per block per K-step, 12 b128 LDS
// loads per 8 WMMAs per wave.
// MODE 0: C = AB                    (QKV, batched over heads; C bf16)
// MODE 1: C = AB + bias + resid     (attention out-proj; C f32)
// MODE 2: C = relu(AB + bias)       (expert FFN layer 1; C bf16)
// MODE 3: C += rowscale[m]*(AB+bias)(expert FFN layer 2, gated; C f32)
// Requires M%256==0, N%64==0, K%32==0 (true for all uses here).
// ---------------------------------------------------------------------------
template <int MODE, typename CT>
__global__ __launch_bounds__(256) void gemm_wmma_kernel(
    const bf16* __restrict__ A, int lda,
    const float* __restrict__ Bm, int ldb, long bstrideB,
    const float* __restrict__ bias,
    CT* __restrict__ C, int ldc, int batch_col_stride,
    const float* __restrict__ resid,
    const float* __restrict__ rowscale, int rs_stride,
    int M, int N, int K) {
  __shared__ bf16 As[256 * 32];   // [m][k]
  __shared__ bf16 BsT[64 * 32];   // [n][k]  (transposed)

  const int tid  = threadIdx.x;
  const int lane = tid & 31;
  const int wave = tid >> 5;      // 0..7 -> 32-row strip
  const int m0   = blockIdx.x * 256;
  const int n0   = blockIdx.y * 64;
  const int z    = blockIdx.z;
  const float* Bz = Bm + (long)z * bstrideB;

  // Staging coordinates
  const int bn  = tid & 63;            // column of B tile
  const int bk8 = (tid >> 6) * 8;      // 8-k group

  v8f acc0[4] = {v8f{}, v8f{}, v8f{}, v8f{}};
  v8f acc1[4] = {v8f{}, v8f{}, v8f{}, v8f{}};

  for (int k0 = 0; k0 < K; k0 += 32) {
    // Stage A (bf16 -> bf16): one full 32-elem row per thread, 4 b128 copies
    // (async global->LDS when available).
    {
      const bf16* ga = A + (long)(m0 + tid) * lda + k0;
      bf16*       la = As + tid * 32;
      async_copy_b128(ga,      la);
      async_copy_b128(ga + 8,  la + 8);
      async_copy_b128(ga + 16, la + 16);
      async_copy_b128(ga + 24, la + 24);
    }
    // Stage B transposed (f32 -> bf16): thread reads 8 k-values of one column
    // (coalesced across lanes at fixed k), packs, one b128 store.
    {
      bf16x8 v;
#pragma unroll
      for (int i = 0; i < 8; ++i)
        v[i] = (bf16)Bz[(long)(k0 + bk8 + i) * ldb + n0 + bn];
      *(bf16x8*)(BsT + bn * 32 + bk8) = v;
    }
    if (k0 + 32 < K)
      __builtin_prefetch(&Bz[(long)(k0 + 32 + bk8) * ldb + n0 + bn], 0, 1);
    async_wait0();       // ASYNCcnt -> 0 before signaling the barrier
    __syncthreads();

    bf16x16 af0 = load_a_frag(&As[(wave * 32) * 32], lane);
    bf16x16 af1 = load_a_frag(&As[(wave * 32 + 16) * 32], lane);
#pragma unroll
    for (int f = 0; f < 4; ++f) {
      bf16x16 bfr = load_bT_frag(BsT, f * 16, lane);
      acc0[f] = wmma_bf16(af0, bfr, acc0[f]);
      acc1[f] = wmma_bf16(af1, bfr, acc1[f]);
    }
    __syncthreads();
  }

  // Epilogue. C frag layout: lane<16 -> rows r, lane>=16 -> rows r+8; col=lane&15.
  const int half = (lane < 16) ? 0 : 8;
  const int nc   = lane & 15;
#pragma unroll
  for (int strip = 0; strip < 2; ++strip) {
#pragma unroll
    for (int f = 0; f < 4; ++f) {
      v8f acc = (strip == 0) ? acc0[f] : acc1[f];
#pragma unroll
      for (int r = 0; r < 8; ++r) {
        int  m  = m0 + wave * 32 + strip * 16 + r + half;
        int  nl = n0 + f * 16 + nc;  // column within batch item
        long ci = (long)m * ldc + (long)z * batch_col_stride + nl;
        float v = acc[r];
        if (MODE != 0) v += bias[nl];
        if (MODE == 2) v = fmaxf(v, 0.0f);
        if (MODE == 1) v += resid[ci];
        if (MODE == 3) {
          C[ci] += rowscale[(long)m * rs_stride] * v;  // unique (m,n); launches serialize
        } else {
          C[ci] = (CT)v;
        }
      }
    }
  }
}

// ---------------------------------------------------------------------------
// Flash attention (causal), bf16 in / bf16 out.
// Q/K/V/O layout: [(b*T + t)*768 + h*64 + dk].
// Grid: (T/64, B*H), block 128 (4 waves); wave owns 16 query rows; block
// iterates shared 32-key tiles up to its causal limit.
// ---------------------------------------------------------------------------
__global__ __launch_bounds__(128) void flash_attn_kernel(
    const bf16* __restrict__ Q, const bf16* __restrict__ Kb,
    const bf16* __restrict__ Vb, bf16* __restrict__ O) {
  __shared__ bf16 Kt[32 * 64];    // [key][dk]
  __shared__ bf16 VtT[64 * 32];   // [dk][key]  (transposed)
  __shared__ bf16 Pb[4][16 * 32];

  const int tid  = threadIdx.x;
  const int lane = tid & 31;
  const int wave = tid >> 5;
  const int bh   = blockIdx.y;
  const int b    = bh / HH;
  const int h    = bh % HH;
  const int q0b  = blockIdx.x * 64;
  const int q0w  = q0b + wave * 16;

  const int  half = (lane < 16) ? 0 : 8;
  const int  nc   = lane & 15;
  const int  ko2  = (lane < 16) ? 0 : 16;
  const long rowbase = (long)b * TT;

  // Load Q fragments once (two contiguous 8-element runs per frag -> b128s).
  bf16x16 qf[2];
  {
    int mq = q0w + (lane & 15);
    int ko = (lane < 16) ? 0 : 16;
    const bf16* qrow = Q + (rowbase + mq) * DD + h * DKK;
#pragma unroll
    for (int s = 0; s < 2; ++s) {
      const bf16x8* p = (const bf16x8*)(qrow + s * 32 + ko);
      bf16x8 lo = p[0];
      bf16x8 hi = p[1];
#pragma unroll
      for (int j = 0; j < 8; ++j) { qf[s][j] = lo[j]; qf[s][j + 8] = hi[j]; }
    }
  }

  v8f o[4] = {v8f{}, v8f{}, v8f{}, v8f{}};
  float rm[8], rl[8];
#pragma unroll
  for (int r = 0; r < 8; ++r) { rm[r] = NEGF; rl[r] = 0.0f; }

  const int kend = q0b + 64;  // causal limit for this block
  for (int kb = 0; kb < kend; kb += 32) {
    // Stage K tile [key][dk]: 16 elems/thread, async b128 copies.
    {
      int r  = tid >> 2;            // 0..31 key
      int cb = (tid & 3) * 16;      // dk group
      const bf16* gk = Kb + (rowbase + kb + r) * DD + h * DKK + cb;
      bf16*       lk = Kt + r * 64 + cb;
      async_copy_b128(gk,     lk);
      async_copy_b128(gk + 8, lk + 8);
    }
    // Stage V transposed [dk][key]: thread gathers 16 keys of one dk column
    // (coalesced across lanes at fixed key), two b128 stores.
    {
      int dk = tid & 63;
      int kg = (tid >> 6) * 16;  // 0 or 16
      bf16x8 lo, hi;
#pragma unroll
      for (int j = 0; j < 8; ++j) {
        lo[j] = Vb[(rowbase + kb + kg + j) * DD + h * DKK + dk];
        hi[j] = Vb[(rowbase + kb + kg + 8 + j) * DD + h * DKK + dk];
      }
      *(bf16x8*)(VtT + dk * 32 + kg)     = lo;
      *(bf16x8*)(VtT + dk * 32 + kg + 8) = hi;
    }
    async_wait0();
    __syncthreads();

    // S = Q @ K^T : two 16x16 score tiles (keys 0-15 / 16-31), 2 WMMAs each.
    v8f s0 = {}, s1 = {};
    {
      const int key0 = nc * 64;
      const int key1 = (16 + nc) * 64;
      bf16x16 bk;
      const bf16x8* p;
      p = (const bf16x8*)(Kt + key0 + ko2);
#pragma unroll
      for (int j = 0; j < 8; ++j) { bk[j] = p[0][j]; bk[j + 8] = p[1][j]; }
      s0 = wmma_bf16(qf[0], bk, s0);
      p = (const bf16x8*)(Kt + key0 + 32 + ko2);
#pragma unroll
      for (int j = 0; j < 8; ++j) { bk[j] = p[0][j]; bk[j + 8] = p[1][j]; }
      s0 = wmma_bf16(qf[1], bk, s0);
      p = (const bf16x8*)(Kt + key1 + ko2);
#pragma unroll
      for (int j = 0; j < 8; ++j) { bk[j] = p[0][j]; bk[j + 8] = p[1][j]; }
      s1 = wmma_bf16(qf[0], bk, s1);
      p = (const bf16x8*)(Kt + key1 + 32 + ko2);
#pragma unroll
      for (int j = 0; j < 8; ++j) { bk[j] = p[0][j]; bk[j + 8] = p[1][j]; }
      s1 = wmma_bf16(qf[1], bk, s1);
    }

    // Online softmax (rows per lane-half; cols across 16 lanes).
#pragma unroll
    for (int r = 0; r < 8; ++r) {
      int   mq = q0w + r + half;
      float a0 = s0[r] * 0.125f;  // 1/sqrt(64)
      float a1 = s1[r] * 0.125f;
      if (kb + nc > mq)      a0 = NEGF;
      if (kb + 16 + nc > mq) a1 = NEGF;
      float mx = fmaxf(a0, a1);
#pragma unroll
      for (int off = 1; off < 16; off <<= 1) mx = fmaxf(mx, __shfl_xor(mx, off, 32));
      float mnew = fmaxf(rm[r], mx);
      float sc   = __expf(rm[r] - mnew);
      float p0   = __expf(a0 - mnew);
      float p1   = __expf(a1 - mnew);
      float ps   = p0 + p1;
#pragma unroll
      for (int off = 1; off < 16; off <<= 1) ps += __shfl_xor(ps, off, 32);
      rl[r] = rl[r] * sc + ps;
      rm[r] = mnew;
      s0[r] = p0;
      s1[r] = p1;
#pragma unroll
      for (int nt = 0; nt < 4; ++nt) o[nt][r] *= sc;
    }

    // Re-fragment P (C-layout -> A-layout) through per-wave LDS.
#pragma unroll
    for (int r = 0; r < 8; ++r) {
      Pb[wave][(r + half) * 32 + nc]      = (bf16)s0[r];
      Pb[wave][(r + half) * 32 + 16 + nc] = (bf16)s1[r];
    }
    __syncthreads();

    bf16x16 pf = load_a_frag(Pb[wave], lane);
#pragma unroll
    for (int nt = 0; nt < 4; ++nt) {
      bf16x16 bv = load_bT_frag(VtT, nt * 16, lane);
      o[nt] = wmma_bf16(pf, bv, o[nt]);
    }
    __syncthreads();
  }

  // Normalize and write O (bf16).
#pragma unroll
  for (int r = 0; r < 8; ++r) {
    int   tq  = q0w + r + half;
    float inv = 1.0f / rl[r];
#pragma unroll
    for (int nt = 0; nt < 4; ++nt)
      O[(rowbase + tq) * DD + h * DKK + nt * 16 + nc] = (bf16)(o[nt][r] * inv);
  }
}

// ---------------------------------------------------------------------------
// RMSNorm: out[row,:] = (bf16)(x * rsqrt(mean(x^2)+eps) * s). Block per row.
// ---------------------------------------------------------------------------
__global__ __launch_bounds__(256) void rmsnorm_kernel(
    const float* __restrict__ x, const float* __restrict__ s,
    bf16* __restrict__ out) {
  __shared__ float red[256];
  const int row = blockIdx.x;
  const int tid = threadIdx.x;
  const float* xr = x + (long)row * DD;
  float ss = 0.0f;
  for (int d = tid; d < DD; d += 256) { float v = xr[d]; ss += v * v; }
  red[tid] = ss;
  __syncthreads();
  for (int o2 = 128; o2 > 0; o2 >>= 1) {
    if (tid < o2) red[tid] += red[tid + o2];
    __syncthreads();
  }
  float rs = rsqrtf(red[0] / (float)DD + 1e-6f);
  for (int d = tid; d < DD; d += 256)
    out[(long)row * DD + d] = (bf16)(xr[d] * rs * s[d]);
}

// ---------------------------------------------------------------------------
// Noisy top-2 gating. One wave per token row; h is bf16.
// ---------------------------------------------------------------------------
__global__ __launch_bounds__(32) void gate_kernel(
    const bf16* __restrict__ h2, const float* __restrict__ wg,
    const float* __restrict__ bg, const float* __restrict__ wv,
    const float* __restrict__ bv, const float* __restrict__ noise,
    float* __restrict__ coef) {
  const int row  = blockIdx.x;
  const int lane = threadIdx.x;
  const bf16* hr = h2 + (long)row * DD;
  float ag[EE] = {}, av[EE] = {};
  for (int d = lane; d < DD; d += 32) {
    float hv = (float)hr[d];
#pragma unroll
    for (int e = 0; e < EE; ++e) {
      ag[e] += hv * wg[d * EE + e];
      av[e] += hv * wv[d * EE + e];
    }
  }
#pragma unroll
  for (int e = 0; e < EE; ++e) {
#pragma unroll
    for (int off = 16; off > 0; off >>= 1) {
      ag[e] += __shfl_xor(ag[e], off, 32);
      av[e] += __shfl_xor(av[e], off, 32);
    }
  }
  if (lane == 0) {
    float lg[EE];
#pragma unroll
    for (int e = 0; e < EE; ++e) {
      float sp = log1pf(__expf(av[e] + bv[e]));  // softplus
      lg[e] = ag[e] + bg[e] + noise[(long)row * EE + e] * sp;
    }
    int i1 = 0, i2 = -1;
    float v1 = lg[0], v2 = NEGF;
#pragma unroll
    for (int e = 1; e < EE; ++e) {
      if (lg[e] > v1) { v2 = v1; i2 = i1; v1 = lg[e]; i1 = e; }
      else if (lg[e] > v2) { v2 = lg[e]; i2 = e; }
    }
    float e2 = __expf(v2 - v1);
    float z  = 1.0f + e2;
#pragma unroll
    for (int e = 0; e < EE; ++e) coef[(long)row * EE + e] = 0.0f;
    coef[(long)row * EE + i1] = 1.0f / z;
    coef[(long)row * EE + i2] = e2 / z;
  }
}

__global__ __launch_bounds__(256) void copy_kernel(
    const float* __restrict__ src, float* __restrict__ dst, long n) {
  long i = (long)blockIdx.x * 256 + threadIdx.x;
  if (i < n) dst[i] = src[i];
}

// ---------------------------------------------------------------------------
// Orchestration
// ---------------------------------------------------------------------------
extern "C" void kernel_launch(void* const* d_in, const int* in_sizes, int n_in,
                              void* d_out, int out_size, void* d_ws, size_t ws_size,
                              hipStream_t stream) {
  (void)in_sizes; (void)n_in; (void)out_size; (void)ws_size;

  const float* x      = (const float*)d_in[0];
  const float* noise  = (const float*)d_in[1];
  const float* wq     = (const float*)d_in[2];   // [H, D, DK]
  const float* wk     = (const float*)d_in[3];
  const float* wv     = (const float*)d_in[4];
  const float* w_proj = (const float*)d_in[5];   // [768, 768]
  const float* b_proj = (const float*)d_in[6];
  const float* w_gate = (const float*)d_in[7];
  const float* b_gate = (const float*)d_in[8];
  const float* w_var  = (const float*)d_in[9];
  const float* b_var  = (const float*)d_in[10];
  const float* w1     = (const float*)d_in[11];  // [E, 768, 3072]
  const float* b1     = (const float*)d_in[12];
  const float* w2     = (const float*)d_in[13];  // [E, 3072, 768]
  const float* b2     = (const float*)d_in[14];
  const float* s1     = (const float*)d_in[15];
  const float* s2     = (const float*)d_in[16];
  float* out = (float*)d_out;

  // Workspace layout (bf16 activations; all chunks 16B-aligned)
  const long SZ = (long)BT_ * DD;           // 3,145,728 elements
  char* w   = (char*)d_ws;
  bf16* h1b = (bf16*)w;                 w += SZ * 2;            // RMSNorm(x)
  bf16* qb  = (bf16*)w;                 w += SZ * 2;
  bf16* kb  = (bf16*)w;                 w += SZ * 2;
  bf16* vb  = (bf16*)w;                 w += SZ * 2;
  bf16* ob  = (bf16*)w;                 w += SZ * 2;            // attention out
  float* x2 = (float*)w;                w += SZ * 4;            // x + proj(o)
  bf16* h2b = (bf16*)w;                 w += SZ * 2;            // RMSNorm(x2)
  float* coef = (float*)w;              w += (long)BT_ * EE * 4;
  bf16* t1b = (bf16*)w;                                         // [BT, DFF]

  dim3 blk256(256);

  // 1) h1 = RMSNorm(x, s1)  (bf16)
  rmsnorm_kernel<<<BT_, 256, 0, stream>>>(x, s1, h1b);

  // 2) Q/K/V: per-head GEMMs batched over z = head (B = wq + z*D*64, D x 64).
  gemm_wmma_kernel<0, bf16><<<dim3(BT_ / 256, 1, HH), blk256, 0, stream>>>(
      h1b, DD, wq, DKK, (long)DD * DKK, nullptr, qb, DD, DKK,
      nullptr, nullptr, 0, BT_, DKK, DD);
  gemm_wmma_kernel<0, bf16><<<dim3(BT_ / 256, 1, HH), blk256, 0, stream>>>(
      h1b, DD, wk, DKK, (long)DD * DKK, nullptr, kb, DD, DKK,
      nullptr, nullptr, 0, BT_, DKK, DD);
  gemm_wmma_kernel<0, bf16><<<dim3(BT_ / 256, 1, HH), blk256, 0, stream>>>(
      h1b, DD, wv, DKK, (long)DD * DKK, nullptr, vb, DD, DKK,
      nullptr, nullptr, 0, BT_, DKK, DD);

  // 3) Causal flash attention -> ob (bf16)
  flash_attn_kernel<<<dim3(TT / 64, BB * HH), 128, 0, stream>>>(qb, kb, vb, ob);

  // 4) x2 = x + ob @ w_proj + b_proj  (f32)
  gemm_wmma_kernel<1, float><<<dim3(BT_ / 256, DD / 64, 1), blk256, 0, stream>>>(
      ob, DD, w_proj, DD, 0, b_proj, x2, DD, 0, x, nullptr, 0, BT_, DD, DD);

  // 5) h2 = RMSNorm(x2, s2)  (bf16)
  rmsnorm_kernel<<<BT_, 256, 0, stream>>>(x2, s2, h2b);

  // 6) gating coefficients
  gate_kernel<<<BT_, 32, 0, stream>>>(h2b, w_gate, b_gate, w_var, b_var, noise, coef);

  // 7) out = x2, then accumulate gated experts
  copy_kernel<<<(unsigned)((SZ + 255) / 256), 256, 0, stream>>>(x2, out, SZ);

  for (int e = 0; e < EE; ++e) {
    // t1 = relu(h2 @ w1[e] + b1[e])  (bf16)
    gemm_wmma_kernel<2, bf16><<<dim3(BT_ / 256, DFF_ / 64, 1), blk256, 0, stream>>>(
        h2b, DD, w1 + (long)e * DD * DFF_, DFF_, 0, b1 + (long)e * DFF_,
        t1b, DFF_, 0, nullptr, nullptr, 0, BT_, DFF_, DD);
    // out += coef[:, e] * (t1 @ w2[e] + b2[e])  (f32 accumulate)
    gemm_wmma_kernel<3, float><<<dim3(BT_ / 256, DD / 64, 1), blk256, 0, stream>>>(
        t1b, DFF_, w2 + (long)e * DFF_ * DD, DD, 0, b2 + (long)e * DD,
        out, DD, 0, nullptr, coef + e, EE, BT_, DD, DFF_);
  }
}
